// LSTMModel_30545807409663
// MI455X (gfx1250) — matrix operand order repaired and checked by reading, not compiled, not run on hardware
//
#include <hip/hip_runtime.h>

// ---------------- problem constants ----------------
#define TT    2514
#define BB    32
#define EMBD  300
#define HIDN  300
#define KPAD  320          // padded K (both EMB and HID) -> 10 chunks of 32
#define NWG   20           // 20 workgroups x 16 hidden cols = 320 padded hidden
#define NKC   (KPAD/32)

// ---------------- workspace layout (bytes) ----------------
#define OFF_CNT   0                      // TT * 4        = 10056
#define OFF_HBUF  10240                  // 2*32*320*2    = 40960  (bf16 h, double buffered)
#define OFF_HMAX  51200                  // 32*320*4      = 40960  (f32 pooled max)
#define OFF_EBF   92160                  // 80448*320*2   = 51486720 (bf16 embedded seq)
#define ZERO_WORDS 12800                 // zero [0, 51200) each call

typedef __attribute__((ext_vector_type(16))) __bf16 v16bf;
typedef __attribute__((ext_vector_type(8)))  float  v8f;

__device__ __forceinline__ unsigned short f2bf(float f) {
  unsigned u = __float_as_uint(f);
  unsigned r = u + 0x7FFFu + ((u >> 16) & 1u);   // round-to-nearest-even
  return (unsigned short)(r >> 16);
}
__device__ __forceinline__ float sigf(float x) { return 1.0f / (1.0f + __expf(-x)); }

// load a 16-element bf16 WMMA fragment from two 16-byte chunks
__device__ __forceinline__ v16bf load16(const unsigned short* p0, const unsigned short* p1) {
  v16bf r;
  ((uint4*)&r)[0] = *(const uint4*)p0;
  ((uint4*)&r)[1] = *(const uint4*)p1;
  return r;
}

// ---------------- kernel 0: zero barrier counters + h double buffer ----------------
__global__ void k_zero(unsigned* p, int n) {
  int i = blockIdx.x * blockDim.x + threadIdx.x;
  if (i < n) p[i] = 0u;
}

// ---------------- kernel 1: embedding gather + f32->bf16, padded to KPAD ----------------
// ebf[t*BB + b][k]  (k in [0,KPAD), zeros for k >= EMBD)
__global__ void k_embed(const int* __restrict__ x, const float* __restrict__ emb,
                        unsigned* __restrict__ ebf_u32, int total_u32) {
  int idx = blockIdx.x * blockDim.x + threadIdx.x;
  if (idx >= total_u32) return;
  int r  = idx / (KPAD / 2);
  int kk = (idx % (KPAD / 2)) * 2;
  int t = r / BB, b = r % BB;
  int tok = x[b * TT + t];
  const float* erow = emb + (size_t)tok * EMBD;
  float e0 = (kk     < EMBD) ? erow[kk]     : 0.0f;
  float e1 = (kk + 1 < EMBD) ? erow[kk + 1] : 0.0f;
  ebf_u32[idx] = (unsigned)f2bf(e0) | ((unsigned)f2bf(e1) << 16);
}

// ---------------- kernel 2: persistent cooperative LSTM scan ----------------
// 20 WGs x 256 threads (8 wave32). WG owns hidden cols [j0, j0+16).
// LDS: W_ih slice(64x320 bf16) + W_hh slice(64x320 bf16) + bias(64f) + gates(4*32*16f) + c(512f) + hmax(512f)
__global__ void k_lstm(const unsigned short* __restrict__ ebf,
                       const float* __restrict__ Wih, const float* __restrict__ Whh,
                       const float* __restrict__ bih, const float* __restrict__ bhh,
                       unsigned short* __restrict__ hbuf, float* __restrict__ hmaxg,
                       unsigned* __restrict__ cnt) {
  extern __shared__ char smem[];
  unsigned short* sWih = (unsigned short*)smem;            // 64*KPAD bf16
  unsigned short* sWhh = sWih + 64 * KPAD;                 // 64*KPAD bf16
  float* sBias = (float*)(sWhh + 64 * KPAD);               // 64
  float* sGate = sBias + 64;                               // 4*512
  float* sC    = sGate + 4 * 512;                          // 512
  float* sHmax = sC + 512;                                 // 512

  const int tid = threadIdx.x;
  const int j0  = blockIdx.x * 16;

  // ---- stage weights (padded with zeros; padded cols self-produce h==0) ----
  for (int idx = tid; idx < 64 * KPAD; idx += 256) {
    int rl = idx / KPAD, k = idx % KPAD;
    int cgl = rl >> 4, rr = rl & 15, j = j0 + rr;
    float wi = 0.0f, wh = 0.0f;
    if (j < HIDN && k < EMBD) wi = Wih[(size_t)(cgl * HIDN + j) * EMBD + k];
    if (j < HIDN && k < HIDN) wh = Whh[(size_t)(cgl * HIDN + j) * HIDN + k];
    sWih[idx] = f2bf(wi);
    sWhh[idx] = f2bf(wh);
  }
  if (tid < 64) {
    int cgl = tid >> 4, rr = tid & 15, j = j0 + rr;
    sBias[tid] = (j < HIDN) ? (bih[cgl * HIDN + j] + bhh[cgl * HIDN + j]) : 0.0f;
  }
  for (int e = tid; e < 512; e += 256) { sC[e] = 0.0f; sHmax[e] = 0.0f; }
  __syncthreads();

  // ---- per-wave tile assignment: wave w -> (M-tile mt, gate class cg) ----
  const int wid  = tid >> 5, lane = tid & 31;
  const int mt   = wid & 1;          // batch tile: rows mt*16..mt*16+15
  const int cg   = wid >> 1;         // 0=i 1=f 2=g 3=o
  const int m    = lane & 15;        // A-row / B-col within tile
  const int klo  = (lane >> 4) << 3; // A layout: lanes 16-31 hold K+8
  const int koff = (lane >> 4) << 4; // unused; B loads use klo, same k order as A

  const unsigned short* wiL = sWih + (size_t)(cg * 16 + m) * KPAD + klo;
  const unsigned short* whL = sWhh + (size_t)(cg * 16 + m) * KPAD + klo;
  const float bv = sBias[cg * 16 + m];

  for (int t = 0; t < TT; ++t) {
    const unsigned short* hcur = hbuf + (size_t)(t & 1) * BB * KPAD;
    unsigned short*       hnxt = hbuf + (size_t)((t + 1) & 1) * BB * KPAD;

    const unsigned short* ea = ebf + ((size_t)t * BB + mt * 16 + m) * KPAD + klo;
    const unsigned short* ha = hcur + (size_t)(mt * 16 + m) * KPAD + klo;
    __builtin_prefetch((const void*)(ea + (size_t)BB * KPAD), 0, 0);  // next timestep

    v8f acc;
    #pragma unroll
    for (int v = 0; v < 8; ++v) acc[v] = bv;

    #pragma unroll
    for (int kc = 0; kc < KPAD; kc += 32) {
      v16bf Ae = load16(ea + kc, ea + kc + 16);
      v16bf Bi = load16(wiL + kc, wiL + kc + 16);
      acc = __builtin_amdgcn_wmma_f32_16x16x32_bf16(false, Ae, false, Bi, (short)0, acc, false, false);
      v16bf Ah = load16(ha + kc, ha + kc + 16);
      v16bf Bh = load16(whL + kc, whL + kc + 16);
      acc = __builtin_amdgcn_wmma_f32_16x16x32_bf16(false, Ah, false, Bh, (short)0, acc, false, false);
    }

    // scatter accumulator to gate staging (C layout: M = v + 8*(lane>=16), N = lane&15)
    {
      const int hi = lane >> 4;
      #pragma unroll
      for (int v = 0; v < 8; ++v) {
        int brow = mt * 16 + v + (hi << 3);
        sGate[cg * 512 + brow * 16 + m] = acc[v];
      }
    }
    __syncthreads();

    // pointwise LSTM cell for this WG's 32x16 slice
    for (int e = tid; e < 512; e += 256) {
      float iv = sigf(sGate[e]);
      float fv = sigf(sGate[512 + e]);
      float gv = tanhf(sGate[1024 + e]);
      float ov = sigf(sGate[1536 + e]);
      float cv = fv * sC[e] + iv * gv;
      sC[e] = cv;
      float hv = ov * tanhf(cv);
      float hr = fmaxf(hv, 0.0f);
      if (hr > sHmax[e]) sHmax[e] = hr;
      int b = e >> 4, n = e & 15;
      hnxt[(size_t)b * KPAD + j0 + n] = f2bf(hv);
    }

    // device-wide barrier (20 WGs): per-step counter, monotonic within the call
    __syncthreads();
    if (tid == 0) {
      __threadfence();
      atomicAdd(&cnt[t], 1u);
      volatile unsigned* vc = (volatile unsigned*)&cnt[t];
      while (*vc < (unsigned)NWG) { __builtin_amdgcn_s_sleep(1); }
      __threadfence();
    }
    __syncthreads();
  }

  // publish pooled max (f32) for this WG's columns
  for (int e = tid; e < 512; e += 256) {
    int b = e >> 4, n = e & 15;
    hmaxg[(size_t)b * KPAD + j0 + n] = sHmax[e];
  }
}

// ---------------- kernel 3: final FC [32x300] @ [300x2] + bias ----------------
__global__ void k_fc(const float* __restrict__ hmaxg, const float* __restrict__ Wfc,
                     const float* __restrict__ bfc, float* __restrict__ out) {
  int tid = threadIdx.x;
  if (tid >= BB * 2) return;
  int b = tid >> 1, n = tid & 1;
  const float* hrow = hmaxg + (size_t)b * KPAD;
  const float* wrow = Wfc + (size_t)n * HIDN;
  float s = 0.0f;
  for (int j = 0; j < HIDN; ++j) s += hrow[j] * wrow[j];
  out[b * 2 + n] = s + bfc[n];
}

// ---------------- launch ----------------
extern "C" void kernel_launch(void* const* d_in, const int* in_sizes, int n_in,
                              void* d_out, int out_size, void* d_ws, size_t ws_size,
                              hipStream_t stream) {
  const int*   x   = (const int*)d_in[0];
  const float* emb = (const float*)d_in[1];
  const float* Wih = (const float*)d_in[2];
  const float* Whh = (const float*)d_in[3];
  const float* bih = (const float*)d_in[4];
  const float* bhh = (const float*)d_in[5];
  const float* Wfc = (const float*)d_in[6];
  const float* bfc = (const float*)d_in[7];
  float* out = (float*)d_out;

  char* ws = (char*)d_ws;
  unsigned*       cnt   = (unsigned*)(ws + OFF_CNT);
  unsigned short* hbuf  = (unsigned short*)(ws + OFF_HBUF);
  float*          hmaxg = (float*)(ws + OFF_HMAX);
  unsigned short* ebf   = (unsigned short*)(ws + OFF_EBF);

  const int smem_bytes = 2 * 64 * KPAD * 2 + (64 + 4 * 512 + 512 + 512) * 4; // 94464
  (void)hipFuncSetAttribute(reinterpret_cast<const void*>(k_lstm),
                            hipFuncAttributeMaxDynamicSharedMemorySize, smem_bytes);

  // 0) zero barrier counters + h double buffer (fresh every call / graph replay)
  k_zero<<<(ZERO_WORDS + 255) / 256, 256, 0, stream>>>((unsigned*)ws, ZERO_WORDS);

  // 1) embedding gather + bf16 convert (fully parallel, off the critical path)
  const int total_u32 = TT * BB * (KPAD / 2);
  k_embed<<<(total_u32 + 255) / 256, 256, 0, stream>>>(x, emb, (unsigned*)ebf, total_u32);

  // 2) cooperative recurrent scan (WMMA bf16, LDS-resident weights)
  k_lstm<<<NWG, 256, smem_bytes, stream>>>(ebf, Wih, Whh, bih, bhh, hbuf, hmaxg, cnt);

  // 3) classifier head
  k_fc<<<1, 64, 0, stream>>>(hmaxg, Wfc, bfc, out);
}
